// FNO3d_44384192037132
// MI455X (gfx1250) — matrix operand order repaired and checked by reading, not compiled, not run on hardware
//
#include <hip/hip_runtime.h>
#include <hip/hip_bf16.h>
#include <math.h>
#include <stdint.h>

// ---------------------------------------------------------------------------
// FNO3d for MI455X (gfx1250, wave32, WMMA + TDM + async-to-LDS).
// Partial-DFT formulation: only the 4*(4x8x8) corner modes are ever computed.
// Heavy GEMMs (skip 1x1 conv, fc1) run on v_wmma_f32_16x16x32_f16 with
// LDS tiles stored in WMMA fragment order (one 32B ds read per fragment).
// ---------------------------------------------------------------------------

typedef __attribute__((ext_vector_type(16))) _Float16 v16h;
typedef __attribute__((ext_vector_type(8)))  float    v8f;
typedef __attribute__((ext_vector_type(4)))  unsigned v4u;
typedef __attribute__((ext_vector_type(8)))  int      v8i;
typedef __attribute__((ext_vector_type(4)))  int      v4i;

#define QS   163840   // T*H*W = 10*128*128
#define HW   16384    // H*W
#define TWO_PI 6.283185307179586f

// ---------------- lift: x[b,o,t,h,w] = lw[o][0]*frame + lw[o][1]*time + lb[o]
__global__ void __launch_bounds__(256) lift_kernel(
    const float* __restrict__ frames, const float* __restrict__ times,
    const float* __restrict__ lw, const float* __restrict__ lb,
    float* __restrict__ x)
{
    const int bo = blockIdx.x;            // b*64 + o
    const int b = bo >> 6, o = bo & 63;
    const int q = blockIdx.y * 256 + threadIdx.x;   // t*HW + h*128 + w
    const int t = q / HW;
    const float f  = frames[(size_t)b * QS + q];
    const float tv = times[b * 10 + t];
    x[(size_t)bo * QS + q] = lw[o * 2] * f + lw[o * 2 + 1] * tv + lb[o];
}

// ---------------- forward partial rDFT along W: LDS-tiled, 32 rows per block
__global__ void __launch_bounds__(256) fwdW_kernel(
    const float* __restrict__ x, float* __restrict__ XW)
{
    __shared__ float tc[128], ts[128];
    __shared__ float rows[32][128];
    const int tid = threadIdx.x;
    if (tid < 128) {
        float a = TWO_PI * tid / 128.f;
        tc[tid] = cosf(a); ts[tid] = sinf(a);
    }
    const size_t row0 = (size_t)blockIdx.x * 32;      // 655360 rows total
    for (int j = tid; j < 4096; j += 256)             // coalesced stage
        rows[j >> 7][j & 127] = x[row0 * 128 + j];
    __syncthreads();
    const int r = tid >> 3, kw = tid & 7;
    float re = 0.f, im = 0.f;
    for (int w = 0; w < 128; ++w) {
        int idx = (kw * w) & 127;
        float v = rows[r][w];
        re += v * tc[idx];
        im -= v * ts[idx];
    }
    const size_t gid = (row0 + r) * 8 + kw;
    XW[gid * 2]     = re;
    XW[gid * 2 + 1] = im;
}

// ---------------- forward partial DFT along H: 128 -> 16 corner freqs
__global__ void __launch_bounds__(256) fwdH_kernel(
    const float* __restrict__ XW, float* __restrict__ XH)
{
    __shared__ float tc[128], ts[128];
    if (threadIdx.x < 128) {
        float a = TWO_PI * threadIdx.x / 128.f;
        tc[threadIdx.x] = cosf(a); ts[threadIdx.x] = sinf(a);
    }
    __syncthreads();
    const int gid = blockIdx.x * 256 + threadIdx.x;   // bct*16*8 = 655360
    const int kw = gid & 7, kh_i = (gid >> 3) & 15, bct = gid >> 7;
    const int kh = (kh_i < 8) ? kh_i : (112 + kh_i);  // {0..7,120..127}
    const float* src = XW + ((size_t)bct * 1024 + kw) * 2;
    float re = 0.f, im = 0.f;
    for (int h = 0; h < 128; ++h) {
        int idx = (kh * h) & 127;
        float xr = src[h * 16], xi = src[h * 16 + 1];
        float cc = tc[idx], ss = ts[idx];
        re += xr * cc + xi * ss;     // * e^{-i theta}
        im += xi * cc - xr * ss;
    }
    XH[(size_t)gid * 2]     = re;
    XH[(size_t)gid * 2 + 1] = im;
}

// ---------------- forward partial DFT along T: 10 -> 8 corner freqs
__global__ void __launch_bounds__(256) fwdT_kernel(
    const float* __restrict__ XH, float* __restrict__ XF)
{
    __shared__ float tc[10], ts[10];
    if (threadIdx.x < 10) {
        float a = TWO_PI * threadIdx.x / 10.f;
        tc[threadIdx.x] = cosf(a); ts[threadIdx.x] = sinf(a);
    }
    __syncthreads();
    const int gid = blockIdx.x * 256 + threadIdx.x;   // 512*1024 = 524288
    const int bc = gid >> 10, f = gid & 1023;
    const int kt_i = f >> 7, kh8 = f & 127;
    const int kt = (kt_i < 4) ? kt_i : (kt_i + 2);    // {0..3,6..9}
    const float* src = XH + ((size_t)bc * 1280 + kh8) * 2;
    float re = 0.f, im = 0.f;
    for (int t = 0; t < 10; ++t) {
        int idx = (kt * t) % 10;
        float xr = src[t * 256], xi = src[t * 256 + 1];
        re += xr * tc[idx] + xi * ts[idx];
        im += xi * tc[idx] - xr * ts[idx];
    }
    XF[(size_t)gid * 2]     = re;
    XF[(size_t)gid * 2 + 1] = im;
}

// ---------------- complex mode multiply: out[b,o,f] = sum_i x[b,i,f]*w[i,o,f]
// Gather of 512 complex values staged with CDNA5 async-to-LDS copies.
__global__ void __launch_bounds__(64) modemul_kernel(
    const float* __restrict__ XF, const float* __restrict__ spec,
    float* __restrict__ XG)
{
    __shared__ float sx[1024];                     // 512 complex (b*64+i)
    const int f = blockIdx.x;                      // 0..1023
    const int kt_i = f >> 7, kh_i = (f >> 3) & 15, kw = f & 7;
    const int blk = ((kt_i >= 4) ? 2 : 0) + ((kh_i >= 8) ? 1 : 0);
    const size_t off = (size_t)blk * 2097152 +
                       (size_t)((kt_i & 3) * 128 + (kh_i & 7) * 16 + kw) * 2;
    // async gather: one b64 (re,im) per lane per iteration
    for (int e = threadIdx.x; e < 512; e += 64) {
        unsigned ldsoff = (unsigned)(uintptr_t)&sx[e * 2];
        const float* g = XF + ((size_t)e * 1024 + f) * 2;
        asm volatile("global_load_async_to_lds_b64 %0, %1, off"
                     :: "v"(ldsoff), "v"(g) : "memory");
    }
    asm volatile("s_wait_asynccnt 0" ::: "memory");
    __syncthreads();
    const int o = threadIdx.x;
    float accr[8], acci[8];
    #pragma unroll
    for (int b = 0; b < 8; ++b) { accr[b] = 0.f; acci[b] = 0.f; }
    for (int i = 0; i < 64; ++i) {
        float wr = spec[off + (size_t)i * 32768 + (size_t)o * 512];
        float wi = spec[off + (size_t)i * 32768 + (size_t)o * 512 + 1];
        #pragma unroll
        for (int b = 0; b < 8; ++b) {
            float xr = sx[(b * 64 + i) * 2], xi = sx[(b * 64 + i) * 2 + 1];
            accr[b] += xr * wr - xi * wi;
            acci[b] += xr * wi + xi * wr;
        }
    }
    #pragma unroll
    for (int b = 0; b < 8; ++b) {
        size_t idx = ((size_t)(b * 64 + o) * 1024 + f) * 2;
        XG[idx]     = accr[b];
        XG[idx + 1] = acci[b];
    }
}

// ---------------- inverse partial DFT along T: 8 corner freqs -> 10
__global__ void __launch_bounds__(256) invT_kernel(
    const float* __restrict__ XG, float* __restrict__ XT)
{
    __shared__ float tc[10], ts[10];
    if (threadIdx.x < 10) {
        float a = TWO_PI * threadIdx.x / 10.f;
        tc[threadIdx.x] = cosf(a); ts[threadIdx.x] = sinf(a);
    }
    __syncthreads();
    const int gid = blockIdx.x * 256 + threadIdx.x;   // 512*10*128 = 655360
    const int bc = gid / 1280, rem = gid % 1280;
    const int t = rem >> 7, kh8 = rem & 127;
    const float* src = XG + ((size_t)bc * 1024 + kh8) * 2;
    float re = 0.f, im = 0.f;
    #pragma unroll
    for (int kt_i = 0; kt_i < 8; ++kt_i) {
        int kt = (kt_i < 4) ? kt_i : (kt_i + 2);
        int idx = (kt * t) % 10;
        float xr = src[kt_i * 256], xi = src[kt_i * 256 + 1];
        re += xr * tc[idx] - xi * ts[idx];   // * e^{+i theta}
        im += xi * tc[idx] + xr * ts[idx];
    }
    XT[(size_t)gid * 2]     = re;
    XT[(size_t)gid * 2 + 1] = im;
}

// ---------------- inverse partial DFT along H: 16 corner freqs -> 128
__global__ void __launch_bounds__(256) invH_kernel(
    const float* __restrict__ XT, float* __restrict__ XHi)
{
    __shared__ float tc[128], ts[128];
    if (threadIdx.x < 128) {
        float a = TWO_PI * threadIdx.x / 128.f;
        tc[threadIdx.x] = cosf(a); ts[threadIdx.x] = sinf(a);
    }
    __syncthreads();
    const int gid = blockIdx.x * 256 + threadIdx.x;   // 5120*128*8 = 5242880
    const int kw = gid & 7, h = (gid >> 3) & 127, bct = gid >> 10;
    const float* src = XT + ((size_t)bct * 128 + kw) * 2;
    float re = 0.f, im = 0.f;
    #pragma unroll
    for (int kh_i = 0; kh_i < 16; ++kh_i) {
        int kh = (kh_i < 8) ? kh_i : (112 + kh_i);
        int idx = (kh * h) & 127;
        float xr = src[kh_i * 16], xi = src[kh_i * 16 + 1];
        re += xr * tc[idx] - xi * ts[idx];
        im += xi * tc[idx] + xr * ts[idx];
    }
    XHi[(size_t)gid * 2]     = re;
    XHi[(size_t)gid * 2 + 1] = im;
}

// ---------------- skip 1x1 conv via WMMA: y[p,o] = sum_i x[p,i]*w[o,i] + b[o]
// skip_w tile DMA'd into LDS by the Tensor Data Mover, then repacked into
// WMMA fragment order so each lane loads its v16h fragment as one 32B read.
__global__ void __launch_bounds__(256) skip_wmma_kernel(
    const float* __restrict__ x, const float* __restrict__ w,
    const float* __restrict__ bias, float* __restrict__ y)
{
    __shared__ float sWf32[64 * 64];                       // raw [o][i] tile
    __shared__ __align__(32) _Float16 sW[2][2][4][16][16]; // [kc][hi][nt][m][e]
    __shared__ __align__(32) _Float16 sA[8][2][16][16];    // [wave][hi][m][e]
    const int tid = threadIdx.x;

#if __has_builtin(__builtin_amdgcn_tensor_load_to_lds)
    if (tid < 32) {   // wave 0 issues the TDM descriptor (EXEC-independent DMA)
        unsigned ldsoff = (unsigned)(uintptr_t)&sWf32[0];
        unsigned long long ga = (unsigned long long)(uintptr_t)w;
        v4u g0 = (v4u){ 1u,                                   // count=1 (valid)
                        ldsoff,                               // lds_addr
                        (unsigned)(ga & 0xffffffffu),         // global_addr lo
                        (unsigned)((ga >> 32) & 0x1ffffffu) | (2u << 30) }; // hi + type=2
        v8i g1 = (v8i){ (int)(2u << 16),          // data_size = 4B
                        (int)(64u << 16),         // tensor_dim0 = 64 (lo16)
                        (int)(64u << 16),         // tensor_dim1 = 64 (lo16)
                        (int)(64u << 16),         // tile_dim0 = 64
                        64,                       // tile_dim1 = 64
                        64,                       // tensor_dim0_stride = 64
                        0, 0 };
        v4i g2 = (v4i){0, 0, 0, 0};
        v4i g3 = (v4i){0, 0, 0, 0};
        v8i g4 = (v8i){0, 0, 0, 0, 0, 0, 0, 0};
        __builtin_amdgcn_tensor_load_to_lds(g0, g1, g2, g3, g4, 0);
        __builtin_amdgcn_s_wait_tensorcnt(0);
    }
#else
    for (int e = tid; e < 4096; e += 256) sWf32[e] = w[e];
#endif
    __syncthreads();
    // repack into WMMA-fragment order (B element e: K = e + 16*hi)
    for (int e4 = tid; e4 < 4096; e4 += 256) {
        int e = e4 & 15, mm = (e4 >> 4) & 15, nt = (e4 >> 8) & 3,
            h2 = (e4 >> 10) & 1, kc = e4 >> 11;
        int i = kc * 32 + h2 * 16 + e;            // K within GEMM
        int o = nt * 16 + mm;                     // N
        sW[kc][h2][nt][mm][e] = (_Float16)sWf32[o * 64 + i];
    }
    __syncthreads();

    const int wv = tid >> 5, lane = tid & 31;
    const int m = lane & 15, hi = lane >> 4;
    const int hiA = lane >> 4;                                 // A staging slot
    const int eA  = lane & 15;
    const long long rt = (long long)blockIdx.x * 8 + wv;       // 16-row tile
    const int b = (int)(rt * 16 / QS);
    const int q = (int)(rt * 16 % QS);
    v8f c[4];
    #pragma unroll
    for (int nt = 0; nt < 4; ++nt) c[nt] = (v8f){};
    #pragma unroll
    for (int kc = 0; kc < 2; ++kc) {
        const int i0 = kc * 32;
        const float* src = x + (size_t)(b * 64 + i0 + lane) * QS + q;
        #pragma unroll
        for (int j = 0; j < 16; ++j)                // lane = K, j = M
            sA[wv][hiA][j][eA] = (_Float16)src[j];  // scatter to fragment order
        const v16h a = *(const v16h*)sA[wv][hi][m];
        #pragma unroll
        for (int nt = 0; nt < 4; ++nt) {
            const v16h bf = *(const v16h*)sW[kc][hi][nt][m];
            c[nt] = __builtin_amdgcn_wmma_f32_16x16x32_f16(
                false, a, false, bf, (short)0, c[nt], false, false);
        }
    }
    #pragma unroll
    for (int nt = 0; nt < 4; ++nt) {
        const int o = nt * 16 + m;
        const float bb = bias[o];
        float* dst = y + (size_t)(b * 64 + o) * QS + q + hi * 8;
        #pragma unroll
        for (int r = 0; r < 8; ++r) dst[r] = c[nt][r] + bb;   // D row M = r+8*hi
    }
}

// ---------------- inverse rDFT along W (8->128 real) + skip add + exact GELU
__global__ void __launch_bounds__(256) invW_gelu_kernel(
    const float* __restrict__ XHi, float* __restrict__ y)
{
    __shared__ float tc[128], ts[128];
    if (threadIdx.x < 128) {
        float a = TWO_PI * threadIdx.x / 128.f;
        tc[threadIdx.x] = cosf(a); ts[threadIdx.x] = sinf(a);
    }
    __syncthreads();
    const size_t gid = (size_t)blockIdx.x * 256 + threadIdx.x;  // 83886080
    const int w = (int)(gid & 127);
    const size_t row = gid >> 7;
    const float* src = XHi + row * 16;
    float s = src[0];                        // kw=0 (real part)
    #pragma unroll
    for (int k = 1; k < 8; ++k) {
        int idx = (k * w) & 127;
        s += 2.f * (src[k * 2] * tc[idx] - src[k * 2 + 1] * ts[idx]);
    }
    s *= (1.f / 163840.f);                   // ortho fwd*inv = 1/(T*H*W)
    float v = s + y[gid];
    y[gid] = 0.5f * v * (1.f + erff(v * 0.70710678118654752f));
}

// ---------------- time aggregation: out[b,o,h,w] = sum_{i,t} x * agg_w + b
__global__ void __launch_bounds__(128) agg_kernel(
    const float* __restrict__ x, const float* __restrict__ aw,
    const float* __restrict__ ab, float* __restrict__ a2)
{
    const int w = threadIdx.x;     // 128
    const int h = blockIdx.x;      // 128
    const int o = blockIdx.y;      // 64
    const int b = blockIdx.z;      // 8
    float acc = ab[o];
    const float* xb = x + (size_t)b * 64 * QS + h * 128 + w;
    const float* wr = aw + o * 640;
    for (int i = 0; i < 64; ++i) {
        const float* xi = xb + (size_t)i * QS;
        #pragma unroll
        for (int t = 0; t < 10; ++t)
            acc += xi[t * HW] * wr[i * 10 + t];
    }
    a2[(size_t)(b * 64 + o) * HW + h * 128 + w] = acc;
}

// ---------------- fc1 via WMMA (K=64) + analytic grid channels + GELU
__global__ void __launch_bounds__(256) fc1_wmma_kernel(
    const float* __restrict__ a2, const float* __restrict__ w,   // (66,128)
    const float* __restrict__ bias, float* __restrict__ fc1o)
{
    __shared__ __align__(32) _Float16 sW[2][2][8][16][16];  // [kc][hi][nt][m][e]
    __shared__ __align__(32) _Float16 sA[8][2][16][16];
    const int tid = threadIdx.x;
    for (int e4 = tid; e4 < 8192; e4 += 256) {
        int e = e4 & 15, mm = (e4 >> 4) & 15, nt = (e4 >> 8) & 7,
            h2 = (e4 >> 11) & 1, kc = e4 >> 12;
        int i = kc * 32 + h2 * 16 + e;
        int n = nt * 16 + mm;
        sW[kc][h2][nt][mm][e] = (_Float16)w[i * 128 + n];
    }
    __syncthreads();
    const int wv = tid >> 5, lane = tid & 31;
    const int m = lane & 15, hi = lane >> 4;
    const int hiA = lane >> 4;
    const int eA  = lane & 15;
    const long long rt = (long long)blockIdx.x * 8 + wv;
    const int b = (int)(rt * 16 / HW);
    const int q = (int)(rt * 16 % HW);
    v8f c[8];
    #pragma unroll
    for (int nt = 0; nt < 8; ++nt) c[nt] = (v8f){};
    #pragma unroll
    for (int kc = 0; kc < 2; ++kc) {
        const int i0 = kc * 32;
        const float* src = a2 + (size_t)(b * 64 + i0 + lane) * HW + q;
        #pragma unroll
        for (int j = 0; j < 16; ++j)
            sA[wv][hiA][j][eA] = (_Float16)src[j];
        const v16h a = *(const v16h*)sA[wv][hi][m];
        #pragma unroll
        for (int nt = 0; nt < 8; ++nt) {
            const v16h bf = *(const v16h*)sW[kc][hi][nt][m];
            c[nt] = __builtin_amdgcn_wmma_f32_16x16x32_f16(
                false, a, false, bf, (short)0, c[nt], false, false);
        }
    }
    #pragma unroll
    for (int nt = 0; nt < 8; ++nt) {
        const int n = nt * 16 + m;
        const float wgx = w[64 * 128 + n], wgy = w[65 * 128 + n], bb = bias[n];
        #pragma unroll
        for (int r = 0; r < 8; ++r) {
            int qq = q + r + hi * 8;
            int h = qq >> 7, ww = qq & 127;
            float gx = -1.f + 2.f * ww / 127.f;
            float gy = -1.f + 2.f * h / 127.f;
            float v = c[nt][r] + gx * wgx + gy * wgy + bb;
            v = 0.5f * v * (1.f + erff(v * 0.70710678118654752f));
            fc1o[(size_t)(b * HW + qq) * 128 + n] = v;
        }
    }
}

// ---------------- fc2 + mu/softplus(sigma) head
__global__ void __launch_bounds__(256) head2_kernel(
    const float* __restrict__ fc1o, const float* __restrict__ w2,
    const float* __restrict__ b2, float* __restrict__ out)
{
    const int p = blockIdx.x * 256 + threadIdx.x;    // 131072
    const float* src = fc1o + (size_t)p * 128;
    float acc[10];
    #pragma unroll
    for (int j = 0; j < 10; ++j) acc[j] = b2[j];
    for (int k = 0; k < 128; ++k) {
        float xv = src[k];
        #pragma unroll
        for (int j = 0; j < 10; ++j) acc[j] += xv * w2[k * 10 + j];
    }
    float* dst = out + (size_t)p * 10;
    #pragma unroll
    for (int u = 0; u < 5; ++u) {
        dst[u] = acc[2 * u];
        float s = acc[2 * u + 1];
        float sp = fmaxf(s, 0.f) + log1pf(expf(-fabsf(s)));   // stable softplus
        dst[5 + u] = sp + 1e-4f;
    }
}

// ---------------------------------------------------------------------------
extern "C" void kernel_launch(void* const* d_in, const int* in_sizes, int n_in,
                              void* d_out, int out_size, void* d_ws, size_t ws_size,
                              hipStream_t stream)
{
    (void)in_sizes; (void)n_in; (void)out_size; (void)ws_size;
    const float* frames = (const float*)d_in[0];
    const float* times  = (const float*)d_in[1];
    const float* lift_w = (const float*)d_in[2];
    const float* lift_b = (const float*)d_in[3];
    const float* spec_w = (const float*)d_in[4];
    const float* skip_w = (const float*)d_in[5];
    const float* skip_b = (const float*)d_in[6];
    const float* agg_w  = (const float*)d_in[7];
    const float* agg_b  = (const float*)d_in[8];
    const float* fc1_w  = (const float*)d_in[9];
    const float* fc1_b  = (const float*)d_in[10];
    const float* fc2_w  = (const float*)d_in[11];
    const float* fc2_b  = (const float*)d_in[12];
    float* out = (float*)d_out;
    float* ws  = (float*)d_ws;

    // workspace layout (floats), aliased to keep footprint ~732 MB
    float* xA = ws;                               // 83,886,080
    float* xB = ws + 83886080ll;                  // 83,886,080
    float* XW = ws + 167772160ll;                 // 10,485,760 (also XHi, agg out)
    float* XH = ws + 178257920ll;                 //  1,310,720
    float* XT = ws + 179568640ll;                 //  1,310,720
    float* XF = ws + 180879360ll;                 //  1,048,576
    float* XG = ws + 181927936ll;                 //  1,048,576
    float* a2   = XW;                             // after spectral layers done
    float* fc1o = xB;                             // xB free after last layer

    lift_kernel<<<dim3(512, 640), 256, 0, stream>>>(frames, times, lift_w, lift_b, xA);

    for (int l = 0; l < 4; ++l) {
        float* xin  = (l & 1) ? xB : xA;
        float* xout = (l & 1) ? xA : xB;
        fwdW_kernel<<<20480, 256, 0, stream>>>(xin, XW);
        fwdH_kernel<<<2560, 256, 0, stream>>>(XW, XH);
        fwdT_kernel<<<2048, 256, 0, stream>>>(XH, XF);
        modemul_kernel<<<1024, 64, 0, stream>>>(XF, spec_w + (size_t)l * 8388608, XG);
        invT_kernel<<<2560, 256, 0, stream>>>(XG, XT);
        invH_kernel<<<20480, 256, 0, stream>>>(XT, XW);     // XHi aliases XW
        skip_wmma_kernel<<<10240, 256, 0, stream>>>(xin, skip_w + l * 4096,
                                                    skip_b + l * 64, xout);
        invW_gelu_kernel<<<327680, 256, 0, stream>>>(XW, xout);
    }

    agg_kernel<<<dim3(128, 64, 8), 128, 0, stream>>>(xA, agg_w, agg_b, a2);
    fc1_wmma_kernel<<<1024, 256, 0, stream>>>(a2, fc1_w, fc1_b, fc1o);
    head2_kernel<<<512, 256, 0, stream>>>(fc1o, fc2_w, fc2_b, out);
}
